// SparseAttention_64802466562673
// MI455X (gfx1250) — compile-verified, matrix operand order checked by reading, not run on hardware
//
#include <hip/hip_runtime.h>
#include <stdint.h>

// ---------------- problem constants ----------------
#define S_   2048
#define D_   1024
#define H_   16
#define DH_  64
#define BLK_ 32
#define NBLK_ 64     // S/BLK
#define NSEL_ 16
#define WIN_ 128
#define CD_  2048    // BLK*DH
#define G3D_ 3072    // 3*D

typedef _Float16 half_t;
typedef __attribute__((ext_vector_type(16))) _Float16 v16h;
typedef __attribute__((ext_vector_type(8)))  _Float16 v8h;
typedef __attribute__((ext_vector_type(8)))  float    v8f;

#define NEGMAX 3.4028235e38f

// ---------------- WMMA helpers (CDNA5 gfx1250, wave32) ----------------
__device__ __forceinline__ v8f wmma16(v16h a, v16h b, v8f c) {
  // D = A(16x32 f16) * B(32x16 f16) + C(16x16 f32)
  return __builtin_amdgcn_wmma_f32_16x16x32_f16(false, a, false, b, (short)0, c,
                                                false, false);
}

// Fragment-ready LDS layout: a 16x32 operand tile is stored as 32 lanes x
// 16 halves (512 halves, 1 KB), so a fragment load is ONE 32-byte LDS read
// (2x ds_load_b128) per lane.
// ISA 7.12.2 inverse map: element (mn, kk) lives at lane = mn | ((kk&8)<<1),
// e = 2*( (kk>=16 ? 4:0) + ((kk&7)>>1) ) + (kk&1).
__device__ __forceinline__ int frag_off(int mn, int kk) {
  const int lane = (mn & 15) | (((kk >> 3) & 1) << 4);
  const int e = ((((kk >> 4) << 2) + ((kk & 7) >> 1)) << 1) + (kk & 1);
  return lane * 16 + e;
}

__device__ __forceinline__ v16h frag_ld(const half_t* tile) {
  return ((const v16h*)tile)[threadIdx.x & 31];
}

// A-fragment gathered from a row-major 16x32 f16 tile in GLOBAL memory
// (row stride ld). K-pairs are contiguous and even -> 8x u32 loads.
__device__ __forceinline__ v16h frag_A_rm(const half_t* p, int ld) {
  const int lane = threadIdx.x & 31;
  const int m = lane & 15;
  const int kb = (lane & 16) ? 8 : 0;
  union { v16h v; uint32_t u[8]; } o;
  const half_t* row = p + m * ld;
#pragma unroll
  for (int r = 0; r < 8; ++r) {
    const int k = (r < 4) ? (kb + 2 * r) : (16 + kb + 2 * (r - 4));
    o.u[r] = *(const uint32_t*)(row + k);
  }
  return o.v;
}

__device__ __forceinline__ v8f v8f_zero() {
  v8f a;
#pragma unroll
  for (int r = 0; r < 8; ++r) a[r] = 0.0f;
  return a;
}

// reductions over the 16 lanes of a wave32 half (xor 1,2,4,8 stays in half)
__device__ __forceinline__ float hmax16(float v) {
  v = fmaxf(v, __shfl_xor(v, 1, 32));
  v = fmaxf(v, __shfl_xor(v, 2, 32));
  v = fmaxf(v, __shfl_xor(v, 4, 32));
  v = fmaxf(v, __shfl_xor(v, 8, 32));
  return v;
}
__device__ __forceinline__ float hsum16(float v) {
  v += __shfl_xor(v, 1, 32);
  v += __shfl_xor(v, 2, 32);
  v += __shfl_xor(v, 4, 32);
  v += __shfl_xor(v, 8, 32);
  return v;
}

// ---------------- RMSNorm -> f16 activations ----------------
__global__ void __launch_bounds__(256) k_rmsnorm(const float* __restrict__ inp,
                                                 const float* __restrict__ nw,
                                                 half_t* __restrict__ xn) {
  const int s = blockIdx.x;
  const int t = threadIdx.x;
  __shared__ float red[8];
  float ss = 0.f;
  for (int i = t; i < D_; i += 256) {
    float v = inp[(size_t)s * D_ + i];
    ss += v * v;
  }
#pragma unroll
  for (int m = 1; m < 32; m <<= 1) ss += __shfl_xor(ss, m, 32);
  if ((t & 31) == 0) red[t >> 5] = ss;
  __syncthreads();
  if (t == 0) {
    float x = 0.f;
#pragma unroll
    for (int w = 0; w < 8; ++w) x += red[w];
    red[0] = rsqrtf(x / (float)D_ + 1.1920929e-7f);
  }
  __syncthreads();
  const float rinv = red[0];
  for (int i = t; i < D_; i += 256)
    xn[(size_t)s * D_ + i] = (half_t)(inp[(size_t)s * D_ + i] * rinv * nw[i]);
}

// ---------------- generic WMMA GEMM: C[M,N] = act(A[M,K] @ W[K,N] + b) ------
// A: f16 row-major.  W: f32 row-major (converted to f16 while staging).
// 256 threads = 8 waves, tile BM=64 x BN=128 x BK=32, 2x4 wave grid,
// 32x32 per wave = 4 WMMA tiles per K-step. LDS tiles in fragment layout.
template <bool RELU>
__global__ void __launch_bounds__(256)
k_gemm(const half_t* __restrict__ A, const float* __restrict__ W,
       const float* __restrict__ bias, float* __restrict__ Cf,
       half_t* __restrict__ Ch, int M, int N, int K) {
  __shared__ __align__(32) half_t sA[4 * 512];   // 4 m-tiles
  __shared__ __align__(32) half_t sB[8 * 512];   // 8 n-tiles
  const int tid = threadIdx.x;
  const int m0 = blockIdx.y * 64;
  const int n0 = blockIdx.x * 128;
  const int wave = tid >> 5, wm = wave >> 2, wn = wave & 3;

  v8f acc[2][2];
  acc[0][0] = v8f_zero(); acc[0][1] = v8f_zero();
  acc[1][0] = v8f_zero(); acc[1][1] = v8f_zero();

  // per-thread staging coordinates
  const int am = tid >> 2;             // 0..63
  const int ak0 = (tid & 3) * 8;       // 0,8,16,24

  for (int k0 = 0; k0 < K; k0 += 32) {
    if (k0 + 32 < K) {  // CDNA5 global_prefetch_b8 for the next K tile
      __builtin_prefetch(&W[(size_t)(k0 + 32) * N + n0], 0, 1);
      __builtin_prefetch(&A[(size_t)m0 * K + k0 + 32], 0, 1);
    }
    __syncthreads();
    {  // A tile: one 16-byte global load, 8 scattered b16 LDS stores
      v8h av = *(const v8h*)&A[(size_t)(m0 + am) * K + k0 + ak0];
      half_t* tile = &sA[(am >> 4) * 512];
#pragma unroll
      for (int q = 0; q < 8; ++q) tile[frag_off(am & 15, ak0 + q)] = av[q];
    }
#pragma unroll
    for (int it = 0; it < 4; ++it) {  // B tile: float4 loads + cvt + scatter
      const int g = tid + it * 256;   // 1024 groups of 4
      const int kk = g >> 5;
      const int c0 = (g & 31) * 4;
      float4 w4 = make_float4(0.f, 0.f, 0.f, 0.f);
      if (n0 + c0 < N) w4 = *(const float4*)&W[(size_t)(k0 + kk) * N + n0 + c0];
      const float wv[4] = {w4.x, w4.y, w4.z, w4.w};
#pragma unroll
      for (int q = 0; q < 4; ++q) {
        const int c = c0 + q;
        sB[(c >> 4) * 512 + frag_off(c & 15, kk)] = (half_t)wv[q];
      }
    }
    __syncthreads();
    v16h a0 = frag_ld(&sA[(wm * 2 + 0) * 512]);
    v16h a1 = frag_ld(&sA[(wm * 2 + 1) * 512]);
#pragma unroll
    for (int tn = 0; tn < 2; ++tn) {
      v16h b = frag_ld(&sB[(wn * 2 + tn) * 512]);
      acc[0][tn] = wmma16(a0, b, acc[0][tn]);
      acc[1][tn] = wmma16(a1, b, acc[1][tn]);
    }
  }

  const int lane = tid & 31;
  const int n_l = lane & 15;
  const int mb8 = (lane >> 4) * 8;
#pragma unroll
  for (int tm = 0; tm < 2; ++tm)
#pragma unroll
    for (int tn = 0; tn < 2; ++tn) {
      const int gn = n0 + wn * 32 + tn * 16 + n_l;
      if (gn >= N) continue;
      const float bv = bias ? bias[gn] : 0.0f;
#pragma unroll
      for (int r = 0; r < 8; ++r) {
        const int gm = m0 + wm * 32 + tm * 16 + mb8 + r;
        float v = acc[tm][tn][r] + bv;
        if (RELU) v = fmaxf(v, 0.0f);
        if (Cf) Cf[(size_t)gm * N + gn] = v;
        if (Ch) Ch[(size_t)gm * N + gn] = (half_t)v;
      }
    }
}

// ---------------- RoPE + head repack + compression inputs ----------------
__global__ void __launch_bounds__(256)
k_prep(const float* __restrict__ qkv, const float* __restrict__ k_pos,
       const float* __restrict__ v_pos, half_t* __restrict__ v_h,
       half_t* __restrict__ qr_h, half_t* __restrict__ kr_h,
       half_t* __restrict__ kcin, half_t* __restrict__ vcin) {
  const int s = blockIdx.x;
  for (int pp = threadIdx.x; pp < D_ / 2; pp += 256) {
    const int d0 = pp * 2;
    const int h = d0 >> 6, dh = d0 & 63;
    const float q0 = qkv[(size_t)s * G3D_ + d0];
    const float q1 = qkv[(size_t)s * G3D_ + d0 + 1];
    const float k0 = qkv[(size_t)s * G3D_ + D_ + d0];
    const float k1 = qkv[(size_t)s * G3D_ + D_ + d0 + 1];
    const float v0 = qkv[(size_t)s * G3D_ + 2 * D_ + d0];
    const float v1 = qkv[(size_t)s * G3D_ + 2 * D_ + d0 + 1];
    const int ip = dh >> 1;  // rotary pair index within head
    const float invf = __expf(-((float)(2 * ip) / (float)DH_) * 9.2103404f);
    float sn, cs;
    __sincosf((float)s * invf, &sn, &cs);
    const size_t hb = ((size_t)h * S_ + s) * DH_ + dh;
    qr_h[hb]     = (half_t)(q0 * cs - q1 * sn);
    qr_h[hb + 1] = (half_t)(q1 * cs + q0 * sn);
    kr_h[hb]     = (half_t)(k0 * cs - k1 * sn);
    kr_h[hb + 1] = (half_t)(k1 * cs + k0 * sn);
    v_h[hb]      = (half_t)v0;
    v_h[hb + 1]  = (half_t)v1;
    const int blk = s >> 5, p = s & 31;
    const size_t crow = ((size_t)(h * NBLK_ + blk)) * CD_ + p * DH_ + dh;
    const size_t pidx = ((size_t)h * BLK_ + p) * DH_ + dh;
    kcin[crow]     = (half_t)(k0 + k_pos[pidx]);
    kcin[crow + 1] = (half_t)(k1 + k_pos[pidx + 1]);
    vcin[crow]     = (half_t)(v0 + v_pos[pidx]);
    vcin[crow + 1] = (half_t)(v1 + v_pos[pidx + 1]);
  }
}

// ---------------- coarse attention over 65 compressed slots ----------------
__global__ void __launch_bounds__(128)
k_coarse(const float* __restrict__ qkv, const float* __restrict__ ck,
         const float* __restrict__ cv, const float* __restrict__ mem_kv,
         float* __restrict__ csim, float* __restrict__ c_out) {
  const int h = blockIdx.x >> 11;
  const int s = blockIdx.x & (S_ - 1);
  const int t = threadIdx.x;
  __shared__ float sim[65];
  __shared__ float snorm;
  const float* qv = qkv + (size_t)s * G3D_ + h * DH_;
  if (t < 65) {
    const float* kv =
        (t == 0) ? (mem_kv + (size_t)h * DH_)
                 : (ck + ((size_t)h * NBLK_ + (t - 1)) * DH_);
    float d = 0.f;
#pragma unroll
    for (int e = 0; e < DH_; ++e) d += qv[e] * kv[e];
    const bool vis = (t == 0) || (t * BLK_ - 1 < s);
    const float v = vis ? d * 0.125f : -NEGMAX;
    sim[t] = v;
    csim[((size_t)h * S_ + s) * 65 + t] = v;
  }
  __syncthreads();
  if (t == 0) {
    float m = -NEGMAX;
    for (int j = 0; j < 65; ++j) m = fmaxf(m, sim[j]);
    float acc = 0.f;
    for (int j = 0; j < 65; ++j) {
      float p = __expf(sim[j] - m);
      sim[j] = p;
      acc += p;
    }
    snorm = 1.0f / acc;
  }
  __syncthreads();
  if (t < DH_) {
    float o = 0.f;
    for (int j = 0; j < 65; ++j) {
      const float* vv =
          (j == 0) ? (mem_kv + ((size_t)H_ + h) * DH_)
                   : (cv + ((size_t)h * NBLK_ + (j - 1)) * DH_);
      o += sim[j] * vv[t];
    }
    c_out[((size_t)h * S_ + s) * DH_ + t] = o * snorm;
  }
}

// ---------------- top-NSEL block selection ----------------
__global__ void __launch_bounds__(256)
k_select(const float* __restrict__ csim, unsigned long long* __restrict__ sel) {
  const int idx = blockIdx.x * blockDim.x + threadIdx.x;  // h*S + s
  if (idx >= H_ * S_) return;
  const float* row = csim + (size_t)idx * 65 + 1;  // 64 block logits
  float m = -1000.0f;  // padded logit
  for (int j = 0; j < NBLK_; ++j) m = fmaxf(m, row[j]);
  float p[NBLK_];
  float acc = __expf(-1000.0f - m);
  for (int j = 0; j < NBLK_; ++j) {
    p[j] = __expf(row[j] - m);
    acc += p[j];
  }
  const float inv = 1.0f / acc;
  unsigned long long mask = 0ull;
  for (int it = 0; it < NSEL_; ++it) {
    int bi = -1;
    float bv = -1.0f;
    for (int j = 0; j < NBLK_; ++j)
      if (!((mask >> j) & 1ull) && p[j] > bv) { bv = p[j]; bi = j; }
    if (bi < 0 || bv * inv <= 1e-10f) break;
    mask |= 1ull << bi;
  }
  sel[idx] = mask;
}

// ---------------- flash-style WMMA attention ----------------
// MODE 0: selected-blocks | own-block-causal mask  -> f_out
// MODE 1: sliding window (i-WIN <= j <= i)         -> s_out
// block = 128 threads (4 waves), each wave owns 16 q rows; kv staged 32 at a
// time in LDS in FRAGMENT layout, shared by all waves (uniform j range).
template <int MODE>
__global__ void __launch_bounds__(128)
k_attn(const half_t* __restrict__ qg, const half_t* __restrict__ kg,
       const half_t* __restrict__ vg, const unsigned long long* __restrict__ sel,
       float* __restrict__ outp) {
  const int h = blockIdx.y;
  const int i0b = blockIdx.x * 64;
  const int wave = threadIdx.x >> 5;
  const int lane = threadIdx.x & 31;
  const int i0w = i0b + wave * 16;
  const int n_l = lane & 15;
  const int mb8 = (lane >> 4) * 8;

  // 4 B^T fragments of K (2 col-tiles x 2 k-slices), 4 B fragments of V
  __shared__ __align__(32) half_t kch[4 * 512];
  __shared__ __align__(32) half_t vch[4 * 512];
  __shared__ __align__(32) half_t pbuf[4][512];  // per-wave A-frag of P

  const half_t* qh = qg + (size_t)h * S_ * DH_;
  const half_t* kh = kg + (size_t)h * S_ * DH_;
  const half_t* vh = vg + (size_t)h * S_ * DH_;

  const v16h qa0 = frag_A_rm(qh + (size_t)i0w * DH_ + 0, DH_);
  const v16h qa1 = frag_A_rm(qh + (size_t)i0w * DH_ + 32, DH_);

  unsigned long long selm[8];
  if (MODE == 0) {
#pragma unroll
    for (int r = 0; r < 8; ++r) selm[r] = sel[(size_t)h * S_ + i0w + mb8 + r];
  }

  v8f acc[4];
#pragma unroll
  for (int t = 0; t < 4; ++t) acc[t] = v8f_zero();
  float rmax[8], rsum[8];
#pragma unroll
  for (int r = 0; r < 8; ++r) { rmax[r] = -NEGMAX; rsum[r] = 0.f; }

  const int js = (MODE == 1 && i0b >= WIN_) ? (i0b - WIN_) : 0;
  const int je = i0b + 64;

  // staging coordinates: 128 threads x 16 halves = 32 rows x 64 dims
  const int scol = threadIdx.x >> 2;        // kv row 0..31
  const int sdim0 = (threadIdx.x & 3) * 16; // dim 0,16,32,48

  for (int j = js; j < je; j += 32) {
    __syncthreads();
    {
      v16h kv = *(const v16h*)&kh[(size_t)(j + scol) * DH_ + sdim0];
      v16h vv = *(const v16h*)&vh[(size_t)(j + scol) * DH_ + sdim0];
#pragma unroll
      for (int q = 0; q < 16; ++q) {
        const int dim = sdim0 + q;
        // K^T fragment: tile (coltile*2 + kslice), B(kk=dim, n=col)
        kch[((scol >> 4) * 2 + (dim >> 5)) * 512 +
            frag_off(scol & 15, dim & 31)] = kv[q];
        // V fragment: tile (dimtile), B(kk=kvrow, n=dim)
        vch[(dim >> 4) * 512 + frag_off(dim & 15, scol)] = vv[q];
      }
    }
    __syncthreads();

    // sim = qr @ kr^T  (two 16x16 tiles covering 32 kv columns)
    v8f s0 = v8f_zero(), s1 = v8f_zero();
    s0 = wmma16(qa0, frag_ld(&kch[0 * 512]), s0);
    s0 = wmma16(qa1, frag_ld(&kch[1 * 512]), s0);
    s1 = wmma16(qa0, frag_ld(&kch[2 * 512]), s1);
    s1 = wmma16(qa1, frag_ld(&kch[3 * 512]), s1);

    float p0[8], p1[8], cf[8];
#pragma unroll
    for (int r = 0; r < 8; ++r) {
      const int i = i0w + mb8 + r;
      const int j0 = j + n_l;
      const int j1 = j + 16 + n_l;
      bool v0, v1;
      if (MODE == 0) {
        v0 = (((selm[r] >> (j0 >> 5)) & 1ull) != 0ull) ||
             (((j0 >> 5) == (i >> 5)) && (j0 <= i));
        v1 = (((selm[r] >> (j1 >> 5)) & 1ull) != 0ull) ||
             (((j1 >> 5) == (i >> 5)) && (j1 <= i));
      } else {
        v0 = (j0 <= i) && (j0 >= i - WIN_);
        v1 = (j1 <= i) && (j1 >= i - WIN_);
      }
      const float sv0 = v0 ? s0[r] * 0.125f : -NEGMAX;
      const float sv1 = v1 ? s1[r] * 0.125f : -NEGMAX;
      float tmax = hmax16(fmaxf(sv0, sv1));
      const float mnew = fmaxf(rmax[r], tmax);
      cf[r] = __expf(rmax[r] - mnew);
      p0[r] = v0 ? __expf(sv0 - mnew) : 0.0f;
      p1[r] = v1 ? __expf(sv1 - mnew) : 0.0f;
      const float ts = hsum16(p0[r] + p1[r]);
      rsum[r] = rsum[r] * cf[r] + ts;
      rmax[r] = mnew;
    }
#pragma unroll
    for (int t = 0; t < 4; ++t)
#pragma unroll
      for (int r = 0; r < 8; ++r) acc[t][r] *= cf[r];

    // write P directly in A-fragment layout (wave-private tile)
    half_t* pb = &pbuf[wave][0];
#pragma unroll
    for (int r = 0; r < 8; ++r) {
      pb[frag_off(mb8 + r, n_l)] = (half_t)p0[r];
      pb[frag_off(mb8 + r, 16 + n_l)] = (half_t)p1[r];
    }
    __syncthreads();
    const v16h pa = frag_ld(pb);
#pragma unroll
    for (int t = 0; t < 4; ++t)
      acc[t] = wmma16(pa, frag_ld(&vch[t * 512]), acc[t]);
  }

#pragma unroll
  for (int t = 0; t < 4; ++t)
#pragma unroll
    for (int r = 0; r < 8; ++r) {
      const int i = i0w + mb8 + r;
      outp[((size_t)h * S_ + i) * DH_ + t * 16 + n_l] = acc[t][r] / rsum[r];
    }
}

// ---------------- gating + combine ----------------
__global__ void __launch_bounds__(256)
k_combine(const float* __restrict__ glin, const float* __restrict__ c_out,
          const float* __restrict__ f_out, const float* __restrict__ s_out,
          half_t* __restrict__ comb) {
  const int s = blockIdx.x;
  for (int i = threadIdx.x; i < D_; i += 256) {
    const int h = i >> 6, d = i & 63;
    const float g0 = 1.0f / (1.0f + __expf(-glin[(size_t)s * (3 * H_) + h * 3 + 0]));
    const float g1 = 1.0f / (1.0f + __expf(-glin[(size_t)s * (3 * H_) + h * 3 + 1]));
    const float g2 = 1.0f / (1.0f + __expf(-glin[(size_t)s * (3 * H_) + h * 3 + 2]));
    const size_t hb = ((size_t)h * S_ + s) * DH_ + d;
    comb[(size_t)s * D_ + i] =
        (half_t)(g0 * c_out[hb] + g1 * f_out[hb] + g2 * s_out[hb]);
  }
}

// ---------------- host launch ----------------
extern "C" void kernel_launch(void* const* d_in, const int* in_sizes, int n_in,
                              void* d_out, int out_size, void* d_ws,
                              size_t ws_size, hipStream_t stream) {
  (void)in_sizes; (void)n_in; (void)out_size; (void)ws_size;
  const float* inp    = (const float*)d_in[0];
  const float* norm_w = (const float*)d_in[1];
  const float* w_qkv  = (const float*)d_in[2];
  const float* k_pos  = (const float*)d_in[3];
  const float* v_pos  = (const float*)d_in[4];
  const float* mem_kv = (const float*)d_in[5];
  const float* kc_w1  = (const float*)d_in[6];
  const float* kc_b1  = (const float*)d_in[7];
  const float* kc_w2  = (const float*)d_in[8];
  const float* kc_b2  = (const float*)d_in[9];
  const float* vc_w1  = (const float*)d_in[10];
  const float* vc_b1  = (const float*)d_in[11];
  const float* vc_w2  = (const float*)d_in[12];
  const float* vc_b2  = (const float*)d_in[13];
  const float* comb_w = (const float*)d_in[14];
  const float* comb_b = (const float*)d_in[15];
  const float* out_w  = (const float*)d_in[16];

  size_t off = 0;
  auto carve = [&](size_t bytes) {
    void* p = (char*)d_ws + off;
    off += (bytes + 255) & ~(size_t)255;
    return p;
  };
  half_t* xn     = (half_t*)carve((size_t)S_ * D_ * 2);
  float*  qkv    = (float*) carve((size_t)S_ * G3D_ * 4);
  half_t* v_h    = (half_t*)carve((size_t)H_ * S_ * DH_ * 2);
  half_t* qr_h   = (half_t*)carve((size_t)H_ * S_ * DH_ * 2);
  half_t* kr_h   = (half_t*)carve((size_t)H_ * S_ * DH_ * 2);
  half_t* kcin   = (half_t*)carve((size_t)H_ * NBLK_ * CD_ * 2);
  half_t* vcin   = (half_t*)carve((size_t)H_ * NBLK_ * CD_ * 2);
  half_t* kc_hid = (half_t*)carve((size_t)H_ * NBLK_ * CD_ * 2);
  half_t* vc_hid = (half_t*)carve((size_t)H_ * NBLK_ * CD_ * 2);
  float*  ck     = (float*) carve((size_t)H_ * NBLK_ * DH_ * 4);
  float*  cv     = (float*) carve((size_t)H_ * NBLK_ * DH_ * 4);
  float*  csim   = (float*) carve((size_t)H_ * S_ * 65 * 4);
  unsigned long long* sel = (unsigned long long*)carve((size_t)H_ * S_ * 8);
  float*  c_out  = (float*) carve((size_t)H_ * S_ * DH_ * 4);
  float*  f_out  = (float*) carve((size_t)H_ * S_ * DH_ * 4);
  float*  s_out  = (float*) carve((size_t)H_ * S_ * DH_ * 4);
  float*  glin   = (float*) carve((size_t)S_ * 3 * H_ * 4);
  half_t* comb   = (half_t*)carve((size_t)S_ * D_ * 2);

  // 1) RMSNorm
  k_rmsnorm<<<S_, 256, 0, stream>>>(inp, norm_w, xn);
  // 2) QKV projection (WMMA)
  k_gemm<false><<<dim3((G3D_ + 127) / 128, S_ / 64), 256, 0, stream>>>(
      xn, w_qkv, nullptr, qkv, nullptr, S_, G3D_, D_);
  // 3) gate linear (WMMA)
  k_gemm<false><<<dim3(1, S_ / 64), 256, 0, stream>>>(
      xn, comb_w, comb_b, glin, nullptr, S_, 3 * H_, D_);
  // 4) RoPE + repack + compression inputs
  k_prep<<<S_, 256, 0, stream>>>(qkv, k_pos, v_pos, v_h, qr_h, kr_h, kcin, vcin);
  // 5) compression MLPs (WMMA)
  k_gemm<true><<<dim3(CD_ / 128, (H_ * NBLK_) / 64), 256, 0, stream>>>(
      kcin, kc_w1, kc_b1, nullptr, kc_hid, H_ * NBLK_, CD_, CD_);
  k_gemm<false><<<dim3(1, (H_ * NBLK_) / 64), 256, 0, stream>>>(
      kc_hid, kc_w2, kc_b2, ck, nullptr, H_ * NBLK_, DH_, CD_);
  k_gemm<true><<<dim3(CD_ / 128, (H_ * NBLK_) / 64), 256, 0, stream>>>(
      vcin, vc_w1, vc_b1, nullptr, vc_hid, H_ * NBLK_, CD_, CD_);
  k_gemm<false><<<dim3(1, (H_ * NBLK_) / 64), 256, 0, stream>>>(
      vc_hid, vc_w2, vc_b2, cv, nullptr, H_ * NBLK_, DH_, CD_);
  // 6) coarse attention + csim
  k_coarse<<<H_ * S_, 128, 0, stream>>>(qkv, ck, cv, mem_kv, csim, c_out);
  // 7) top-k block selection
  k_select<<<(H_ * S_) / 256, 256, 0, stream>>>(csim, sel);
  // 8) selected-block attention (WMMA flash)
  k_attn<0><<<dim3(S_ / 64, H_), 128, 0, stream>>>(qr_h, kr_h, v_h, sel, f_out);
  // 9) sliding-window attention (WMMA flash)
  k_attn<1><<<dim3(S_ / 64, H_), 128, 0, stream>>>(qr_h, kr_h, v_h, nullptr, s_out);
  // 10) gate + combine
  k_combine<<<S_, 256, 0, stream>>>(glin, c_out, f_out, s_out, comb);
  // 11) output projection (WMMA) -> d_out
  k_gemm<false><<<dim3(D_ / 128, S_ / 64), 256, 0, stream>>>(
      comb, out_w, nullptr, (float*)d_out, nullptr, S_, D_, D_);
}
